// GraphAttentionLayer_39599598469573
// MI455X (gfx1250) — matrix operand order repaired and checked by reading, not compiled, not run on hardware
//
#include <hip/hip_runtime.h>

// ---------------------------------------------------------------------------
// GAT layer, fused for MI455X (gfx1250, wave32, WMMA + TDM).
//   N=8192 nodes, 256 in/out features, LeakyReLU(0.2), +/- masked softmax.
// Rank-1 scores: e_ij = lrelu(s1_i + s2_j); never materialize [N,N].
// Heavy matmuls via v_wmma_f32_16x16x32_f16 fed from ping-pong LDS tiles.
// B tiles (pre-converted f16, feature-major) are moved by the Tensor Data
// Mover (tensor_load_to_lds + s_wait_tensorcnt); issue is guarded by a
// readfirstlane-uniform branch so exactly ONE wave programs the TDM.
// ---------------------------------------------------------------------------

#define NROW 8192
#define NF   256
#define ALPHA 0.2f
#define INV_N (1.0f / 8192.0f)

typedef __attribute__((ext_vector_type(16))) _Float16 v16h;
typedef __attribute__((ext_vector_type(8)))  float    v8f;
typedef __attribute__((ext_vector_type(4)))  float    f4;
typedef __attribute__((ext_vector_type(4)))  int      i4;
typedef __attribute__((ext_vector_type(4)))  unsigned u4;
typedef __attribute__((ext_vector_type(8)))  int      tdm8i;
typedef __attribute__((ext_vector_type(4)))  int      tdm4i;
typedef __attribute__((ext_vector_type(4)))  unsigned tdm4u;

// --- TDM availability / arity detection -----------------------------------
#if defined(__has_builtin)
#  if __has_builtin(__builtin_amdgcn_tensor_load_to_lds) && \
      __has_builtin(__builtin_amdgcn_s_wait_tensorcnt)
#    define USE_TDM 1
#  else
#    define USE_TDM 0
#  endif
#else
#  define USE_TDM 0
#endif

#if defined(__has_include)
#  if __has_include(<hip/amd_detail/amd_gfx1250_TDM.h>)
#    define TDM_SIX_ARG 1
#  else
#    define TDM_SIX_ARG 0
#  endif
#else
#  define TDM_SIX_ARG 0
#endif

__device__ __forceinline__ float lrelu(float x) { return x >= 0.f ? x : ALPHA * x; }

#if USE_TDM
// 2D tile load: tile_d0 x tile_d1 elements (2-byte), row stride stride0,
// from global gaddr into LDS offset lds_off.  D# layout per CDNA5 ISA ch.8.
__device__ __forceinline__ void tdm_load_2d(unsigned lds_off, const void* gaddr,
                                            unsigned tile_d0, unsigned tile_d1,
                                            unsigned tensor_d0, unsigned tensor_d1,
                                            unsigned stride0) {
    unsigned long long ga = (unsigned long long)(size_t)gaddr;
    tdm4u g0;
    g0[0] = 1u;                                                // count=1 (valid), user mode
    g0[1] = lds_off;                                           // lds_addr (bytes)
    g0[2] = (unsigned)(ga & 0xffffffffu);                      // global_addr[31:0]
    g0[3] = (unsigned)((ga >> 32) & 0x01ffffffu) | (2u << 30); // addr[56:32] | type=2
    tdm8i g1;
    g1[0] = (int)(1u << 16);                                   // wg_mask=0, data_size=1 (2B)
    g1[1] = (int)(tensor_d0 << 16);                            // tensor_dim0[15:0]
    g1[2] = (int)((tensor_d0 >> 16) | (tensor_d1 << 16));      // dim0 hi | tensor_dim1 lo
    g1[3] = (int)((tensor_d1 >> 16) | (tile_d0 << 16));        // dim1 hi | tile_dim0
    g1[4] = (int)(tile_d1 & 0xffffu);                          // tile_dim1, tile_dim2=0
    g1[5] = (int)stride0;                                      // tensor_dim0_stride[31:0]
    g1[6] = 0;
    g1[7] = 0;
    tdm4i gz = {0, 0, 0, 0};
#if TDM_SIX_ARG
    tdm8i gz8 = {0, 0, 0, 0, 0, 0, 0, 0};
    __builtin_amdgcn_tensor_load_to_lds(g0, g1, gz, gz, gz8, 0);
#else
    __builtin_amdgcn_tensor_load_to_lds(g0, g1, gz, gz, 0);
#endif
}
#endif

// ---------------------------------------------------------------- k_prep ----
__global__ void k_prep(const float* __restrict__ W, const float* __restrict__ a,
                       float* __restrict__ b) {
    int f = threadIdx.x;
    float acc1 = 0.f, acc2 = 0.f;
    for (int o = 0; o < NF; ++o) {
        float w = W[f * NF + o];
        acc1 += w * a[o];
        acc2 += w * a[NF + o];
    }
    b[f] = acc1;
    b[256 + f] = acc2;
}

// -------------------------------------------------------------- k_scores ----
__global__ void k_scores(const float* __restrict__ inp, const float* __restrict__ b,
                         float* __restrict__ s1, float* __restrict__ s2) {
    int wave = threadIdx.x >> 5, lane = threadIdx.x & 31;
    int i = blockIdx.x * 8 + wave;
    const float* row = inp + (size_t)i * NF;
    float p1 = 0.f, p2 = 0.f;
    for (int q = 0; q < NF / 32; ++q) {
        float x = row[lane + 32 * q];
        p1 += x * b[lane + 32 * q];
        p2 += x * b[256 + lane + 32 * q];
    }
    for (int off = 16; off > 0; off >>= 1) {
        p1 += __shfl_xor(p1, off, 32);
        p2 += __shfl_xor(p2, off, 32);
    }
    if (lane == 0) { s1[i] = p1; s2[i] = p2; }
}

// -------------------------------------------------------------- k_cvt_inp ---
__global__ void k_cvt_inp(const float* __restrict__ inp, _Float16* __restrict__ inpT) {
    __shared__ __align__(16) _Float16 T[256][32];
    int tid = threadIdx.x;
    int j0 = blockIdx.x * 32;
    {
        int k = tid >> 3, c0 = (tid & 7) * 32;
        const f4* src = (const f4*)(inp + (size_t)(j0 + k) * NF + c0);
#pragma unroll
        for (int q = 0; q < 8; ++q) {
            f4 v = src[q];
            int f = c0 + q * 4;
            T[f + 0][k] = (_Float16)v.x; T[f + 1][k] = (_Float16)v.y;
            T[f + 2][k] = (_Float16)v.z; T[f + 3][k] = (_Float16)v.w;
        }
    }
    __syncthreads();
    int f = tid >> 1, h = tid & 1;
#pragma unroll
    for (int rep = 0; rep < 2; ++rep, f += 128) {
        const u4* s = (const u4*)&T[f][0];
        u4* d = (u4*)(inpT + (size_t)f * NROW + j0);
        d[h * 2 + 0] = s[h * 2 + 0];
        d[h * 2 + 1] = s[h * 2 + 1];
    }
}

// -------------------------------------------------------------- k_cvt_wt ----
__global__ void k_cvt_wt(const float* __restrict__ wtrans, _Float16* __restrict__ wtT) {
    __shared__ __align__(16) _Float16 T[256][32];
    int tid = threadIdx.x;
    int k0 = blockIdx.x * 32;
    {
        int k = tid >> 3, c0 = (tid & 7) * 32;
        const f4* src = (const f4*)(wtrans + (size_t)(k0 + k) * NF + c0);
#pragma unroll
        for (int q = 0; q < 8; ++q) {
            f4 v = src[q];
            int f = c0 + q * 4;
            T[f + 0][k] = (_Float16)v.x; T[f + 1][k] = (_Float16)v.y;
            T[f + 2][k] = (_Float16)v.z; T[f + 3][k] = (_Float16)v.w;
        }
    }
    __syncthreads();
    int o = tid >> 1, h = tid & 1;
#pragma unroll
    for (int rep = 0; rep < 2; ++rep, o += 128) {
        const u4* s = (const u4*)&T[o][0];
        u4* d = (u4*)(wtT + (size_t)o * 512 + k0);
        d[h * 2 + 0] = s[h * 2 + 0];
        d[h * 2 + 1] = s[h * 2 + 1];
    }
}

// --------------------------------------------------------------- k_stats ----
__global__ void k_stats(const int* __restrict__ adj,
                        const float* __restrict__ s1, const float* __restrict__ s2,
                        float* __restrict__ mP, float* __restrict__ zP,
                        float* __restrict__ mN, float* __restrict__ zN,
                        float* __restrict__ unif) {
    __shared__ float red[256];
    __shared__ float sMax, sMin, sZp;
    int i = blockIdx.x, t = threadIdx.x;
    const i4* row4 = (const i4*)(adj + (size_t)i * NROW);

    float mx = -3.4e38f, mn = 3.4e38f;
    for (int q = 0; q < NROW / 4 / 256; ++q) {
        i4 av = row4[t + 256 * q];
        int j = (t + 256 * q) * 4;
#pragma unroll
        for (int e = 0; e < 4; ++e)
            if (av[e] > 0) { float v = s2[j + e]; mx = fmaxf(mx, v); mn = fminf(mn, v); }
    }
    red[t] = mx; __syncthreads();
    for (int s = 128; s > 0; s >>= 1) { if (t < s) red[t] = fmaxf(red[t], red[t + s]); __syncthreads(); }
    if (t == 0) sMax = red[0]; __syncthreads();
    red[t] = mn; __syncthreads();
    for (int s = 128; s > 0; s >>= 1) { if (t < s) red[t] = fminf(red[t], red[t + s]); __syncthreads(); }
    if (t == 0) sMin = red[0]; __syncthreads();

    float S1 = s1[i];
    bool empty = (sMax < -1e37f);
    float mp = empty ? 0.f : lrelu(S1 + sMax);
    float mq = empty ? 0.f : -lrelu(S1 + sMin);

    float zp = 0.f, zn = 0.f;
    for (int q = 0; q < NROW / 4 / 256; ++q) {
        i4 av = row4[t + 256 * q];
        int j = (t + 256 * q) * 4;
#pragma unroll
        for (int e = 0; e < 4; ++e)
            if (av[e] > 0) {
                float ev = lrelu(S1 + s2[j + e]);
                zp += __expf(ev - mp);
                zn += __expf(-ev - mq);
            }
    }
    red[t] = zp; __syncthreads();
    for (int s = 128; s > 0; s >>= 1) { if (t < s) red[t] += red[t + s]; __syncthreads(); }
    if (t == 0) sZp = red[0]; __syncthreads();
    red[t] = zn; __syncthreads();
    for (int s = 128; s > 0; s >>= 1) { if (t < s) red[t] += red[t + s]; __syncthreads(); }
    if (t == 0) {
        mP[i] = mp; mN[i] = mq;
        zP[i] = empty ? 0.f : 1.f / sZp;
        zN[i] = empty ? 0.f : 1.f / red[0];
        unif[i] = empty ? 1.f : 0.f;
    }
}

// ---------------------------------------------------------------- k_main ----
// Block = 16 i-rows, 8 waves.  Ping-pong LDS.  TDM pattern per chunk:
//   wave0: s_wait_tensorcnt(0)      (Bt[p] landed)
//   all:   stage As[p] weights
//   all:   barrier                  (publishes Bt[p]+As[p]; frees Bt[p^1])
//   wave0: tensor_load_to_lds  ->  Bt[p^1]   (overlaps with WMMA below)
//   all:   WMMA on As[p] x Bt[p]
// wave0 is identified via readfirstlane so the condition is SGPR-uniform and
// the backend emits a real scalar branch (TENSOR ops ignore EXEC, so an
// exec-predicated path would issue the DMA from every wave).
__global__ __launch_bounds__(256) void k_main(
    const _Float16* __restrict__ inpT, const int* __restrict__ adj,
    const _Float16* __restrict__ wtT,
    const float* __restrict__ s1, const float* __restrict__ s2,
    const float* __restrict__ mP, const float* __restrict__ zP,
    const float* __restrict__ mN, const float* __restrict__ zN,
    const float* __restrict__ unif, float* __restrict__ out) {

    __shared__ __align__(16) _Float16 As[2][2][16][32];   // [buf][sign][row][col]
    __shared__ __align__(16) _Float16 Bt[2][256][32];     // [buf][feature][k]
    __shared__ __align__(16) _Float16 Hc[16][512];        // concat [hp | -hn]

    int tid = threadIdx.x;
    int wave = tid >> 5, lane = tid & 31;
    int hi = lane >> 4;
    int mrow = lane & 15;
    int ncol = lane & 15;
    int fbase = wave * 32;
    int i0 = blockIdx.x * 16;

    int c  = tid & 31;
    int r0 = tid >> 5;
    int ia = i0 + r0, ib = i0 + r0 + 8;
    float S1a = s1[ia], MPa = mP[ia], ZPa = zP[ia], MNa = mN[ia], ZNa = zN[ia], Ua = unif[ia];
    float S1b = s1[ib], MPb = mP[ib], ZPb = zP[ib], MNb = mN[ib], ZNb = zN[ib], Ub = unif[ib];
    const size_t adjA = (size_t)ia * NROW;
    const size_t adjB = (size_t)ib * NROW;

    int cf = tid >> 1, ch = tid & 1;   // (fallback copy roles)
    (void)cf; (void)ch;

#if USE_TDM
    // SGPR-uniform "am I wave 0" flag -> scalar branch around TDM ops.
    const bool w0 = (__builtin_amdgcn_readfirstlane(wave) == 0);
    const unsigned bt0 = (unsigned)(size_t)&Bt[0][0][0];
    const unsigned bt1 = (unsigned)(size_t)&Bt[1][0][0];
    if (w0)  // preload first inp tile into Bt[0]
        tdm_load_2d(bt0, inpT, 32, 256, NROW, 256, NROW);
#endif

    v8f accP0 = {}, accP1 = {}, accN0 = {}, accN1 = {};

    for (int j0 = 0; j0 < NROW; j0 += 32) {
        int p = (j0 >> 5) & 1;
        int j = j0 + c;

#if USE_TDM
        if (w0) __builtin_amdgcn_s_wait_tensorcnt(0);
#endif
        // prefetch the adj stream (only HBM-bound stream, 268MB)
        if (j + 1024 < NROW) {
            __builtin_prefetch(&adj[adjA + j + 1024], 0, 1);
            __builtin_prefetch(&adj[adjB + j + 1024], 0, 1);
        }
        // --- branchless attention weights for rows r0, r0+8 at column c ---
        {
            float s2j = s2[j];
            int av0 = adj[adjA + j];
            int av1 = adj[adjB + j];
            float e0 = lrelu(S1a + s2j);
            float e1 = lrelu(S1b + s2j);
            float m0 = av0 > 0 ? 1.f : 0.f;
            float m1 = av1 > 0 ? 1.f : 0.f;
            float wp0 = m0 * __expf(e0 - MPa) * ZPa;  wp0 += Ua * (INV_N - wp0);
            float wn0 = m0 * __expf(-e0 - MNa) * ZNa; wn0 += Ua * (INV_N - wn0);
            float wp1 = m1 * __expf(e1 - MPb) * ZPb;  wp1 += Ub * (INV_N - wp1);
            float wn1 = m1 * __expf(-e1 - MNb) * ZNb; wn1 += Ub * (INV_N - wn1);
            As[p][0][r0][c]     = (_Float16)wp0;
            As[p][0][r0 + 8][c] = (_Float16)wp1;
            As[p][1][r0][c]     = (_Float16)wn0;
            As[p][1][r0 + 8][c] = (_Float16)wn1;
        }
#if !USE_TDM
        {
#pragma unroll
            for (int rep = 0; rep < 2; ++rep) {
                int f = cf + rep * 128;
                const u4* s = (const u4*)(inpT + (size_t)f * NROW + j0);
                u4* d = (u4*)&Bt[p][f][0];
                d[ch * 2 + 0] = s[ch * 2 + 0];
                d[ch * 2 + 1] = s[ch * 2 + 1];
            }
        }
#endif
        __syncthreads();

#if USE_TDM
        if (w0 && j0 + 32 < NROW)   // DMA next tile into the freed buffer
            tdm_load_2d(p ? bt0 : bt1, inpT + (j0 + 32), 32, 256, NROW, 256, NROW);
#endif

        // --- fragments + WMMA ---
        union { v16h h; unsigned u[8]; } aP, aN;
        {
            const unsigned* bp = (const unsigned*)&As[p][0][mrow][0];
            const unsigned* bn = (const unsigned*)&As[p][1][mrow][0];
            int o = hi * 4;
#pragma unroll
            for (int v = 0; v < 4; ++v) { aP.u[v] = bp[o + v];         aN.u[v] = bn[o + v]; }
#pragma unroll
            for (int v = 0; v < 4; ++v) { aP.u[4 + v] = bp[8 + o + v]; aN.u[4 + v] = bn[8 + o + v]; }
        }
        union { v16h h; u4 q[2]; } b0, b1;
        {
            const u4* p0 = (const u4*)&Bt[p][fbase + ncol][hi * 8];
            b0.q[0] = p0[0]; b0.q[1] = p0[2];
            const u4* p1 = (const u4*)&Bt[p][fbase + 16 + ncol][hi * 8];
            b1.q[0] = p1[0]; b1.q[1] = p1[2];
        }
        accP0 = __builtin_amdgcn_wmma_f32_16x16x32_f16(false, aP.h, false, b0.h, (short)0, accP0, false, false);
        accP1 = __builtin_amdgcn_wmma_f32_16x16x32_f16(false, aP.h, false, b1.h, (short)0, accP1, false, false);
        accN0 = __builtin_amdgcn_wmma_f32_16x16x32_f16(false, aN.h, false, b0.h, (short)0, accN0, false, false);
        accN1 = __builtin_amdgcn_wmma_f32_16x16x32_f16(false, aN.h, false, b1.h, (short)0, accN1, false, false);
    }
    __syncthreads();   // all j-loop LDS reads complete

#if USE_TDM
    if (w0)  // preload first wtrans tile (overlaps with Hc stores)
        tdm_load_2d(bt0, wtT, 32, 256, 512, 256, 512);
#endif

    // --- concat tile Hc[m][0:256]=hp, Hc[m][256:512]=-hn ---
#pragma unroll
    for (int v = 0; v < 8; ++v) {
        int M = v + 8 * hi;
        Hc[M][fbase + ncol]            = (_Float16)accP0[v];
        Hc[M][fbase + 16 + ncol]       = (_Float16)accP1[v];
        Hc[M][256 + fbase + ncol]      = (_Float16)(-accN0[v]);
        Hc[M][256 + fbase + 16 + ncol] = (_Float16)(-accN1[v]);
    }
    __syncthreads();

    // --- projection: out = elu(Hc @ wtrans), K=512 ---
    v8f d0 = {}, d1 = {};
    int obase = wave * 32;
    for (int k0 = 0; k0 < 512; k0 += 32) {
        int p = (k0 >> 5) & 1;
#if USE_TDM
        if (w0) __builtin_amdgcn_s_wait_tensorcnt(0);
        __syncthreads();
        if (w0 && k0 + 32 < 512)
            tdm_load_2d(p ? bt0 : bt1, wtT + (k0 + 32), 32, 256, 512, 256, 512);
#else
        {
#pragma unroll
            for (int rep = 0; rep < 2; ++rep) {
                int o = cf + rep * 128;
                const u4* s = (const u4*)(wtT + (size_t)o * 512 + k0);
                u4* d = (u4*)&Bt[p][o][0];
                d[ch * 2 + 0] = s[ch * 2 + 0];
                d[ch * 2 + 1] = s[ch * 2 + 1];
            }
        }
        __syncthreads();
#endif

        union { v16h h; unsigned u[8]; } ah;
        {
            const unsigned* bp = (const unsigned*)&Hc[mrow][k0];
            int o = hi * 4;
#pragma unroll
            for (int v = 0; v < 4; ++v) ah.u[v] = bp[o + v];
#pragma unroll
            for (int v = 0; v < 4; ++v) ah.u[4 + v] = bp[8 + o + v];
        }
        union { v16h h; u4 q[2]; } bb0, bb1;
        {
            const u4* p0 = (const u4*)&Bt[p][obase + ncol][hi * 8];
            bb0.q[0] = p0[0]; bb0.q[1] = p0[2];
            const u4* p1 = (const u4*)&Bt[p][obase + 16 + ncol][hi * 8];
            bb1.q[0] = p1[0]; bb1.q[1] = p1[2];
        }
        d0 = __builtin_amdgcn_wmma_f32_16x16x32_f16(false, ah.h, false, bb0.h, (short)0, d0, false, false);
        d1 = __builtin_amdgcn_wmma_f32_16x16x32_f16(false, ah.h, false, bb1.h, (short)0, d1, false, false);
    }

    // --- ELU + store ---
#pragma unroll
    for (int v = 0; v < 8; ++v) {
        int M = v + 8 * hi;
        float x0 = d0[v]; x0 = x0 > 0.f ? x0 : (__expf(x0) - 1.f);
        float x1 = d1[v]; x1 = x1 > 0.f ? x1 : (__expf(x1) - 1.f);
        out[(size_t)(i0 + M) * NF + obase + ncol]      = x0;
        out[(size_t)(i0 + M) * NF + obase + 16 + ncol] = x1;
    }
}

// ---------------------------------------------------------------------------
extern "C" void kernel_launch(void* const* d_in, const int* in_sizes, int n_in,
                              void* d_out, int out_size, void* d_ws, size_t ws_size,
                              hipStream_t stream) {
    const float* inp    = (const float*)d_in[0];
    const int*   adj    = (const int*)d_in[1];
    const float* W      = (const float*)d_in[2];
    const float* a      = (const float*)d_in[3];
    const float* wtrans = (const float*)d_in[4];
    float* out = (float*)d_out;

    float* ws = (float*)d_ws;
    float* b  = ws;
    float* s1 = ws + 512;
    float* s2 = s1 + NROW;
    float* mP = s2 + NROW;
    float* zP = mP + NROW;
    float* mN = zP + NROW;
    float* zN = mN + NROW;
    float* un = zN + NROW;
    _Float16* inpT = (_Float16*)((char*)d_ws + (256u << 10));              // 4MB
    _Float16* wtT  = (_Float16*)((char*)d_ws + (256u << 10) + (4u << 20)); // 256KB

    k_prep   <<<1,         256, 0, stream>>>(W, a, b);
    k_scores <<<NROW / 8,  256, 0, stream>>>(inp, b, s1, s2);
    k_cvt_inp<<<NROW / 32, 256, 0, stream>>>(inp, inpT);
    k_cvt_wt <<<512 / 32,  256, 0, stream>>>(wtrans, wtT);
    k_stats  <<<NROW,      256, 0, stream>>>(adj, s1, s2, mP, zP, mN, zN, un);
    k_main   <<<NROW / 16, 256, 0, stream>>>(inpT, adj, wtT, s1, s2,
                                             mP, zP, mN, zN, un, out);
}
